// variationalseq2seq_54047868453333
// MI455X (gfx1250) — matrix operand order repaired and checked by reading, not compiled, not run on hardware
//
#include <hip/hip_runtime.h>
#include <hip/hip_bf16.h>

typedef __bf16 bf16;
typedef __attribute__((ext_vector_type(16))) __bf16 v16bf;
typedef __attribute__((ext_vector_type(8)))  __bf16 v8bf;
typedef __attribute__((ext_vector_type(8)))  float  v8f;

#define H_ 512
#define E_ 300
#define EP 320          // E padded to multiple of 32 (WMMA K)
#define B_ 32
#define S_ 400
#define T_ 15
#define V_ 50000
#define VP 50048        // V padded to multiple of 128
#define G4 2048         // 4*H

__device__ __forceinline__ v8f wmma_bf16(v16bf a, v16bf b, v8f c) {
  // D(16x16 f32) = A(16x32 bf16) * B(32x16 bf16) + C
  return __builtin_amdgcn_wmma_f32_16x16x32_bf16(false, a, false, b, (short)0, c,
                                                 false, false);
}

__device__ __forceinline__ v8f vzero8() {
  v8f z;
#pragma unroll
  for (int i = 0; i < 8; ++i) z[i] = 0.f;
  return z;
}

// A-fragment, 16x32 bf16 (MxK), row-major source.
// ISA layout: lane L<16 holds M=L, K in {half*8..half*8+7} u {16+half*8..16+half*8+7}
__device__ __forceinline__ v16bf load_a_frag(const bf16* base, int ldk, int k0, int lane) {
  int half = (lane >> 4) & 1;
  int m = lane & 15;
  const bf16* p = base + m * ldk + k0 + half * 8;
  v8bf lo = *(const v8bf*)p;
  v8bf hi = *(const v8bf*)(p + 16);
  v16bf a;
#pragma unroll
  for (int i = 0; i < 8; ++i) { a[i] = lo[i]; a[i + 8] = hi[i]; }
  return a;
}

__device__ __forceinline__ float sigm(float x) { return 1.f / (1.f + expf(-x)); }

// ---------------------------------------------------------------------------
// Persistent LSTM kernel. grid = 2 blocks (dir 0 fwd, dir 1 bwd), 1024 threads
// (32 waves). Gate matrix [32 x 2048] tiled 2(M) x 128(N); wave w: m-tile w&1,
// n-tiles (w>>1)*8 .. +7. Weight columns gate-interleaved: n = 4*j + gate.
// ---------------------------------------------------------------------------
__global__ void __launch_bounds__(1024)
lstm_kernel(const bf16* __restrict__ xseq, int Kx,
            const bf16* __restrict__ Wt,      // [2][(Kx+H)][2048] K-major bf16
            const float* __restrict__ bias,   // [2][4H] original gate-major
            bf16* __restrict__ oout,          // optional [steps][32][2H], cols dir*H..
            float* __restrict__ hfin, float* __restrict__ cfin,   // enc finals [2][32][H]
            bf16* __restrict__ hstate, float* __restrict__ cstate, // dec persist [2][32][H]
            int steps, int reverse, int enc_mode)
{
  __shared__ __align__(64) bf16  hbuf[B_ * H_];   // 32 KB
  __shared__ __align__(64) float cbuf[B_ * H_];   // 64 KB
  const int dir  = blockIdx.x;
  const int tid  = threadIdx.x;
  const int lane = tid & 31;
  const int wave = tid >> 5;
  const int m0   = (wave & 1) * 16;
  const int wn   = wave >> 1;
  const int half = (lane >> 4) & 1;
  const int nlo  = lane & 15;
  const int Kcat = Kx + H_;
  const bf16*  Wd = Wt + (size_t)dir * Kcat * G4 + wn * 128;  // fold n-tile base
  const float* bd = bias + dir * G4;

  int colj[8]; int colg[8]; float brg[8];
#pragma unroll
  for (int ti = 0; ti < 8; ++ti) {
    int col = wn * 128 + ti * 16 + nlo;
    colj[ti] = col >> 2; colg[ti] = col & 3;
    brg[ti]  = bd[colg[ti] * H_ + colj[ti]];
  }

  if (enc_mode) {
    for (int i = tid; i < B_ * H_; i += blockDim.x) { hbuf[i] = (bf16)0.f; cbuf[i] = 0.f; }
  } else {
    const bf16*  hs = hstate + (size_t)dir * B_ * H_;
    const float* cs = cstate + (size_t)dir * B_ * H_;
    for (int i = tid; i < B_ * H_; i += blockDim.x) { hbuf[i] = hs[i]; cbuf[i] = cs[i]; }
  }
  __syncthreads();

  for (int t = 0; t < steps; ++t) {
    int tt = (reverse && dir == 1) ? (steps - 1 - t) : t;
    const bf16* xrow = xseq + (size_t)tt * B_ * Kx + m0 * Kx;
    v8f acc[8];
#pragma unroll
    for (int ti = 0; ti < 8; ++ti) acc[ti] = vzero8();

    // ---- x-region: A from global, straight-line loop ----
    for (int k0 = 0; k0 < Kx; k0 += 32) {
      const bf16* wrow = Wd + ((size_t)k0 + half * 8) * G4 + nlo;
      v16bf bf[8];
#pragma unroll
      for (int ti = 0; ti < 8; ++ti) for (int s = 0; s < 16; ++s) bf[ti][s] = wrow[(size_t)(s + (s & 8)) * G4 + ti * 16];
      v16bf a = load_a_frag(xrow, Kx, k0, lane);
      __builtin_prefetch(wrow + (size_t)32 * G4, 0, 0);
#pragma unroll
      for (int ti = 0; ti < 8; ++ti) acc[ti] = wmma_bf16(a, bf[ti], acc[ti]);
    }
    // ---- h-region: A from LDS ----
    {
      const bf16* Wh = Wd + (size_t)Kx * G4;
      const bf16* hrow = hbuf + m0 * H_;
      for (int k0 = 0; k0 < H_; k0 += 32) {
        const bf16* wrow = Wh + ((size_t)k0 + half * 8) * G4 + nlo;
        v16bf bf[8];
#pragma unroll
        for (int ti = 0; ti < 8; ++ti) for (int s = 0; s < 16; ++s) bf[ti][s] = wrow[(size_t)(s + (s & 8)) * G4 + ti * 16];
        v16bf a = load_a_frag(hrow, H_, k0, lane);
#pragma unroll
        for (int ti = 0; ti < 8; ++ti) acc[ti] = wmma_bf16(a, bf[ti], acc[ti]);
      }
    }
    __syncthreads();   // all hbuf reads for this step complete

#pragma unroll
    for (int ti = 0; ti < 8; ++ti) {
      int j = colj[ti], g = colg[ti];
#pragma unroll
      for (int r = 0; r < 8; ++r) {
        float p = acc[ti][r] + brg[ti];
        float act = (g == 2) ? tanhf(p) : sigm(p);
        int lb = lane & ~3;
        float ai = __shfl(act, lb + 0, 32);
        float af = __shfl(act, lb + 1, 32);
        float ag = __shfl(act, lb + 2, 32);
        float ao = __shfl(act, lb + 3, 32);
        if ((lane & 3) == 0) {
          int brow = m0 + r + 8 * half;
          float c = af * cbuf[brow * H_ + j] + ai * ag;
          cbuf[brow * H_ + j] = c;
          float h = ao * tanhf(c);
          hbuf[brow * H_ + j] = (bf16)h;
          if (oout)
            oout[(size_t)tt * B_ * (2 * H_) + brow * (2 * H_) + dir * H_ + j] = (bf16)h;
        }
      }
    }
    __syncthreads();   // h visible before next step's reads
  }

  if (enc_mode) {
    float* hf = hfin + (size_t)dir * B_ * H_;
    float* cf = cfin + (size_t)dir * B_ * H_;
    for (int i = tid; i < B_ * H_; i += blockDim.x) { hf[i] = (float)hbuf[i]; cf[i] = cbuf[i]; }
  } else {
    bf16*  hs = hstate + (size_t)dir * B_ * H_;
    float* cs = cstate + (size_t)dir * B_ * H_;
    for (int i = tid; i < B_ * H_; i += blockDim.x) { hs[i] = hbuf[i]; cs[i] = cbuf[i]; }
  }
}

// ---------------------------------------------------------------------------
// Prediction GEMM: [32 x 1024] x [1024 x VP] -> logits f32. 391 blocks x 8 waves,
// each wave one 16-column tile for both M tiles.
// ---------------------------------------------------------------------------
__global__ void __launch_bounds__(256)
pred_kernel(const bf16* __restrict__ A, const bf16* __restrict__ Wt,
            const float* __restrict__ bias, float* __restrict__ logits)
{
  const int lane = threadIdx.x & 31;
  const int wave = threadIdx.x >> 5;
  const int n0   = blockIdx.x * 128 + wave * 16;
  const int half = (lane >> 4) & 1;
  v8f acc0 = vzero8(), acc1 = vzero8();
#pragma unroll 4
  for (int k0 = 0; k0 < 2 * H_; k0 += 32) {
    v16bf b;
    for (int s = 0; s < 16; ++s) b[s] = Wt[((size_t)k0 + half * 8 + s + (s & 8)) * VP + n0 + (lane & 15)];
    v16bf a0 = load_a_frag(A, 2 * H_, k0, lane);
    v16bf a1 = load_a_frag(A + 16 * 2 * H_, 2 * H_, k0, lane);
    acc0 = wmma_bf16(a0, b, acc0);
    acc1 = wmma_bf16(a1, b, acc1);
  }
  int col = n0 + (lane & 15);
  float bv = (col < V_) ? bias[col] : 0.f;
#pragma unroll
  for (int r = 0; r < 8; ++r) {
    int row = r + 8 * half;
    logits[(size_t)row * VP + col]        = acc0[r] + bv;
    logits[(size_t)(row + 16) * VP + col] = acc1[r] + bv;
  }
}

// Per-batch-row log_softmax + argmax. grid = 32 blocks.
__global__ void __launch_bounds__(256)
softmax_kernel(const float* __restrict__ logits, float* __restrict__ outp,
               int* __restrict__ tok)
{
  int b = blockIdx.x, tid = threadIdx.x;
  const float* row = logits + (size_t)b * VP;
  __shared__ float sv[256]; __shared__ int si[256];
  float bm = -1e30f; int bi = 0;
  for (int v = tid; v < V_; v += 256) { float x = row[v]; if (x > bm) { bm = x; bi = v; } }
  sv[tid] = bm; si[tid] = bi; __syncthreads();
  for (int s = 128; s > 0; s >>= 1) {
    if (tid < s) {
      if (sv[tid + s] > sv[tid] || (sv[tid + s] == sv[tid] && si[tid + s] < si[tid])) {
        sv[tid] = sv[tid + s]; si[tid] = si[tid + s];
      }
    }
    __syncthreads();
  }
  float maxv = sv[0]; int arg = si[0]; __syncthreads();
  float ls = 0.f;
  for (int v = tid; v < V_; v += 256) ls += expf(row[v] - maxv);
  sv[tid] = ls; __syncthreads();
  for (int s = 128; s > 0; s >>= 1) { if (tid < s) sv[tid] += sv[tid + s]; __syncthreads(); }
  float logZ = maxv + logf(sv[0]);
  for (int v = tid; v < V_; v += 256) outp[(size_t)b * V_ + v] = row[v] - logZ;
  if (tid == 0) tok[b] = arg;
}

// ---------------------------------------------------------------------------
// Prep: transpose f32 weight [R][C] -> bf16 K-major [Kpad][Npad], optional
// gate-interleave perm (dst col n <- src row (n&3)*512 + (n>>2)).
// ---------------------------------------------------------------------------
__global__ void wt_kernel(const float* __restrict__ W, bf16* __restrict__ dst,
                          int Rw, int Cw, int Kpad, int Npad, int perm)
{
  size_t idx = (size_t)blockIdx.x * blockDim.x + threadIdx.x;
  if (idx >= (size_t)Kpad * Npad) return;
  int k = (int)(idx / Npad), n = (int)(idx % Npad);
  int r = perm ? ((n & 3) * H_ + (n >> 2)) : n;
  float v = (k < Cw && r < Rw) ? W[(size_t)r * Cw + k] : 0.f;
  dst[idx] = (bf16)v;
}

__global__ void embA_kernel(const int* __restrict__ toks, const float* __restrict__ emb,
                            bf16* __restrict__ x, int n)
{
  int idx = blockIdx.x * blockDim.x + threadIdx.x;
  if (idx >= n * EP) return;
  int sb = idx / EP, e = idx % EP;
  int tk = toks[sb];
  x[idx] = (e < E_) ? (bf16)emb[(size_t)tk * E_ + e] : (bf16)0.f;
}

__global__ void dec_embed_kernel(const int* __restrict__ summary0, const int* __restrict__ tok,
                                 int t, const float* __restrict__ emb, bf16* __restrict__ x)
{
  int idx = blockIdx.x * blockDim.x + threadIdx.x;
  if (idx >= B_ * EP) return;
  int b = idx / EP, e = idx % EP;
  int tk = (t == 0) ? summary0[b] : tok[b];
  x[idx] = (e < E_) ? (bf16)emb[(size_t)tk * E_ + e] : (bf16)0.f;
}

__global__ void vae1_kernel(const float* __restrict__ hid, const float* __restrict__ muW,
                            const float* __restrict__ mub, const float* __restrict__ lvW,
                            const float* __restrict__ lvb, const float* __restrict__ zn,
                            float* __restrict__ mu_o, float* __restrict__ lv_o,
                            float* __restrict__ z_o)
{
  int idx = blockIdx.x * blockDim.x + threadIdx.x;
  if (idx >= 4 * B_ * H_) return;
  int j = idx & (H_ - 1); int row = idx >> 9;
  const float* hr = hid + (size_t)row * H_;
  float sm = mub[j], sl = lvb[j];
  for (int k = 0; k < H_; ++k) {
    float hv = hr[k];
    sm += hv * muW[(size_t)j * H_ + k];
    sl += hv * lvW[(size_t)j * H_ + k];
  }
  float z = zn[(row & (B_ - 1)) * H_ + j] * expf(0.5f * sl) + sm;
  mu_o[idx] = sm; lv_o[idx] = sl; z_o[idx] = z;
}

__global__ void vae2_kernel(const float* __restrict__ z, const float* __restrict__ hW,
                            const float* __restrict__ hb, const float* __restrict__ cfin,
                            bf16* __restrict__ hstate, float* __restrict__ cstate)
{
  int idx = blockIdx.x * blockDim.x + threadIdx.x;
  if (idx >= 4 * B_ * H_) return;
  int j = idx & (H_ - 1); int row = idx >> 9;
  const float* zr = z + (size_t)row * H_;
  float s = hb[j];
  for (int k = 0; k < H_; ++k) s += zr[k] * hW[(size_t)j * H_ + k];
  hstate[idx] = (bf16)s;
  cstate[idx] = cfin[idx];
}

// ---------------------------------------------------------------------------
extern "C" void kernel_launch(void* const* d_in, const int* in_sizes, int n_in,
                              void* d_out, int out_size, void* d_ws, size_t ws_size,
                              hipStream_t stream)
{
  (void)in_sizes; (void)n_in; (void)out_size; (void)ws_size;
  const int*   article = (const int*)d_in[0];
  const int*   summary = (const int*)d_in[1];
  const float* z_noise = (const float*)d_in[2];
  const float* enc_emb = (const float*)d_in[3];
  const float* e0Wih = (const float*)d_in[4];
  const float* e0Whh = (const float*)d_in[5];
  const float* e0b   = (const float*)d_in[6];
  const float* e1Wih = (const float*)d_in[7];
  const float* e1Whh = (const float*)d_in[8];
  const float* e1b   = (const float*)d_in[9];
  const float* muW = (const float*)d_in[10];
  const float* mub = (const float*)d_in[11];
  const float* lvW = (const float*)d_in[12];
  const float* lvb = (const float*)d_in[13];
  const float* dhW = (const float*)d_in[14];
  const float* dhb = (const float*)d_in[15];
  const float* dec_emb = (const float*)d_in[16];
  const float* d0Wih = (const float*)d_in[17];
  const float* d0Whh = (const float*)d_in[18];
  const float* d0b   = (const float*)d_in[19];
  const float* d1Wih = (const float*)d_in[20];
  const float* d1Whh = (const float*)d_in[21];
  const float* d1b   = (const float*)d_in[22];
  const float* predW = (const float*)d_in[23];
  const float* predb = (const float*)d_in[24];

  float* outp = (float*)d_out;
  float* mu_o = outp + (size_t)T_ * B_ * V_;
  float* lv_o = mu_o + 4 * B_ * H_;
  float* z_o  = lv_o + 4 * B_ * H_;

  char* wptr = (char*)d_ws;
  auto alloc = [&](size_t bytes) -> void* {
    void* p = (void*)wptr; wptr += (bytes + 255) & ~(size_t)255; return p;
  };
  bf16*  WtE0 = (bf16*)alloc((size_t)2 * (EP + H_) * G4 * 2);
  bf16*  WtE1 = (bf16*)alloc((size_t)2 * (3 * H_) * G4 * 2);
  bf16*  WtD0 = (bf16*)alloc((size_t)2 * (EP + H_) * G4 * 2);
  bf16*  WtD1 = (bf16*)alloc((size_t)2 * (3 * H_) * G4 * 2);
  bf16*  WtP  = (bf16*)alloc((size_t)2 * H_ * VP * 2);
  bf16*  embB = (bf16*)alloc((size_t)S_ * B_ * EP * 2);
  bf16*  o0   = (bf16*)alloc((size_t)S_ * B_ * 2 * H_ * 2);
  float* hfin = (float*)alloc((size_t)4 * B_ * H_ * 4);
  float* cfin = (float*)alloc((size_t)4 * B_ * H_ * 4);
  bf16*  dh   = (bf16*)alloc((size_t)4 * B_ * H_ * 2);
  float* dc   = (float*)alloc((size_t)4 * B_ * H_ * 4);
  bf16*  decx = (bf16*)alloc((size_t)B_ * EP * 2);
  bf16*  dx1  = (bf16*)alloc((size_t)B_ * 2 * H_ * 2);
  bf16*  dh1  = (bf16*)alloc((size_t)B_ * 2 * H_ * 2);
  float* logits = (float*)alloc((size_t)B_ * VP * 4);
  int*   tok    = (int*)alloc(B_ * 4);

  auto wt = [&](const float* W, bf16* dst, int Rw, int Cw, int Kpad, int Npad, int perm) {
    size_t tot = (size_t)Kpad * Npad;
    wt_kernel<<<(unsigned)((tot + 255) / 256), 256, 0, stream>>>(W, dst, Rw, Cw, Kpad, Npad, perm);
  };

  for (int d = 0; d < 2; ++d) {
    bf16* p0 = WtE0 + (size_t)d * (EP + H_) * G4;
    wt(e0Wih + (size_t)d * G4 * E_,     p0,                     G4, E_,     EP,     G4, 1);
    wt(e0Whh + (size_t)d * G4 * H_,     p0 + (size_t)EP * G4,   G4, H_,     H_,     G4, 1);
    bf16* p1 = WtE1 + (size_t)d * (3 * H_) * G4;
    wt(e1Wih + (size_t)d * G4 * 2 * H_, p1,                     G4, 2 * H_, 2 * H_, G4, 1);
    wt(e1Whh + (size_t)d * G4 * H_,     p1 + (size_t)2*H_*G4,   G4, H_,     H_,     G4, 1);
    bf16* q0 = WtD0 + (size_t)d * (EP + H_) * G4;
    wt(d0Wih + (size_t)d * G4 * E_,     q0,                     G4, E_,     EP,     G4, 1);
    wt(d0Whh + (size_t)d * G4 * H_,     q0 + (size_t)EP * G4,   G4, H_,     H_,     G4, 1);
    bf16* q1 = WtD1 + (size_t)d * (3 * H_) * G4;
    wt(d1Wih + (size_t)d * G4 * 2 * H_, q1,                     G4, 2 * H_, 2 * H_, G4, 1);
    wt(d1Whh + (size_t)d * G4 * H_,     q1 + (size_t)2*H_*G4,   G4, H_,     H_,     G4, 1);
  }
  wt(predW, WtP, V_, 2 * H_, 2 * H_, VP, 0);
  embA_kernel<<<(S_ * B_ * EP + 255) / 256, 256, 0, stream>>>(article, enc_emb, embB, S_ * B_);

  // ---- encoder: 2-layer bidirectional LSTM (persistent, one block per dir) ----
  lstm_kernel<<<2, 1024, 0, stream>>>(embB, EP, WtE0, e0b, o0,
                                      hfin, cfin, nullptr, nullptr, S_, 1, 1);
  lstm_kernel<<<2, 1024, 0, stream>>>(o0, 2 * H_, WtE1, e1b, nullptr,
                                      hfin + 2 * B_ * H_, cfin + 2 * B_ * H_,
                                      nullptr, nullptr, S_, 1, 1);

  // ---- VAE reparameterization + decoder init ----
  vae1_kernel<<<(4 * B_ * H_ + 255) / 256, 256, 0, stream>>>(hfin, muW, mub, lvW, lvb,
                                                             z_noise, mu_o, lv_o, z_o);
  vae2_kernel<<<(4 * B_ * H_ + 255) / 256, 256, 0, stream>>>(z_o, dhW, dhb, cfin, dh, dc);

  // ---- decoder: greedy argmax feedback ----
  for (int t = 0; t < T_; ++t) {
    dec_embed_kernel<<<(B_ * EP + 255) / 256, 256, 0, stream>>>(summary, tok, t, dec_emb, decx);
    lstm_kernel<<<2, 1024, 0, stream>>>(decx, EP, WtD0, d0b, dx1,
                                        nullptr, nullptr, dh, dc, 1, 0, 0);
    lstm_kernel<<<2, 1024, 0, stream>>>(dx1, 2 * H_, WtD1, d1b, dh1,
                                        nullptr, nullptr, dh + 2 * B_ * H_, dc + 2 * B_ * H_,
                                        1, 0, 0);
    pred_kernel<<<VP / 128, 256, 0, stream>>>(dh1, WtP, predb, logits);
    softmax_kernel<<<B_, 256, 0, stream>>>(logits, outp + (size_t)t * B_ * V_, tok);
  }
}
